// FFTResonanceLookup_67310727463593
// MI455X (gfx1250) — matrix operand order repaired and checked by reading, not compiled, not run on hardware
//
#include <hip/hip_runtime.h>
#include <hip/hip_bf16.h>

typedef _Float16 f16;
typedef __attribute__((ext_vector_type(16))) _Float16 v16h;
typedef __attribute__((ext_vector_type(8)))  _Float16 v8h;
typedef __attribute__((ext_vector_type(8)))  float    v8f;

#define PI_F 3.14159265358979f

#define N_ROWS      512      // 8*16*4 batch rows
#define K_SEL       512      // inner dim of selection GEMM
#define N_COEFFS    771
#define N_COEFFS_PAD 784     // 49*16
#define CHUNK       257
#define NFRAMES     128
#define WINDOW      512
#define STEP        256
#define K2V         514      // 2*CHUNK (real/imag interleaved)
#define K2PAD       544      // 17*32
#define OUT_PER_ROW 32768

// ---- workspace layout (bytes, all offsets 256B aligned) ----
#define OFF_AF16   0u                      // 512*512*2   = 524288
#define OFF_IT     524288u                 // 784*512*2   = 802816
#define OFF_C1     (524288u+802816u)       // 512*784*4   = 1605632
#define OFF_BT     (2932736u)              // 512*544*2   = 557056
#define OFF_PART   (3489792u)              // 4096*4      = 16384
#define OFF_NORM   (3506176u)              // 512*4

// ======================= prep kernels =======================

// relu(selections) -> f16, row-major [512][512]
__global__ void k_prep_a(const float* __restrict__ sel, f16* __restrict__ A) {
    int i = blockIdx.x * 256 + threadIdx.x;
    float v = sel[i];
    A[i] = (f16)(v > 0.f ? v : 0.f);
}

// items[k][n] -> It[n][k] f16, n padded to 784 (zeros)
__global__ void k_prep_items(const float* __restrict__ items, f16* __restrict__ It) {
    int n = blockIdx.x;          // 0..783
    int k = threadIdx.x;         // 0..511
    f16 v = (f16)0.f;
    if (n < N_COEFFS) v = (f16)items[k * N_COEFFS + n];
    It[n * K_SEL + k] = v;
}

// DFT basis (transposed, n-major): Bt[n][k2], k2 even -> w*h*cos, odd -> -w*h*sin
// hann and irfft weights folded in; k2 >= 514 zero padded.
__global__ void k_prep_basis(f16* __restrict__ Bt) {
    int n  = blockIdx.x;         // 0..511
    int k2 = threadIdx.x;        // 0..543
    int k  = k2 >> 1;
    float val = 0.f;
    if (k2 < K2V) {
        float w = ((k == 0) || (k == 256)) ? (1.f / 512.f) : (2.f / 512.f);
        float h = 0.5f - 0.5f * __cosf(2.f * PI_F * (float)n / 511.f);  // np.hanning(512)
        int kn = (k * n) & 511;                                         // exact period-512 reduction
        float th = 2.f * PI_F * (float)kn * (1.f / 512.f);
        float s, c;
        __sincosf(th, &s, &c);
        if (k2 & 1) val = ((k == 0) || (k == 256)) ? 0.f : -w * h * s;  // imag of DC/Nyquist ignored
        else        val = w * h * c;
    }
    Bt[n * K2PAD + k2] = (f16)val;
}

// ======================= stage 1: selection GEMM =======================
// C1[512][784] = relu(sel)[512][512] @ items[512][784], f16 WMMA, one wave per 16x16 tile.
__global__ void k_gemm1(const f16* __restrict__ A, const f16* __restrict__ It,
                        float* __restrict__ C1) {
    int tile  = blockIdx.x;          // 0..1567
    int mtile = tile / 49;
    int ntile = tile % 49;
    int lane = threadIdx.x;          // 0..31
    int m = lane & 15;
    int h = lane >> 4;

    const f16* Arow = A  + (mtile * 16 + m) * K_SEL;      // A operand: row = M = lane&15
    const f16* Brow = It + (ntile * 16 + m) * K_SEL;      // B operand: col = N = lane&15
    v8f acc = {};
    #pragma unroll 4
    for (int kt = 0; kt < 16; ++kt) {
        int kb = kt * 32;
        v16h a, b;
        a = *(const v16h*)(Arow + kb + 16 * h);            // K = kb + 16h + j
        b = *(const v16h*)(Brow + kb + 16 * h);            // K = kb + 16h + j
        acc = __builtin_amdgcn_wmma_f32_16x16x32_f16(false, a, false, b, (short)0, acc, false, false);
    }
    #pragma unroll
    for (int r = 0; r < 8; ++r) {
        int row = mtile * 16 + r + 8 * h;                  // D: M = r + 8*half
        int col = ntile * 16 + m;                          // D: N = lane&15
        C1[row * N_COEFFS_PAD + col] = acc[r];
    }
}

// ======================= stage 2: fused resonance synth =======================
// One workgroup per (row r, frame-tile ft). Owns output segments [16ft..16ft+15].
// Phase 1: nonlinearities + spec generation -> S[17][544] f16 in LDS (row 0 = boundary frame).
// Phase 2: 16x512 frame tile via WMMA (K=544); A fragment loaded once per kt and
//          reused across 4 N-tile accumulators. Boundary frame 2nd half via VALU dot.
// Phase 3: overlap-add, store unnormalized output, sumsq partial.
__global__ void __launch_bounds__(256) k_synth(const float* __restrict__ C1,
                                               const f16* __restrict__ Bt,
                                               float* __restrict__ out,
                                               float* __restrict__ partial) {
    __shared__ __align__(32) f16 S[17][K2PAD];   // 18496 B
    __shared__ float F[16][WINDOW];              // 32768 B
    __shared__ float extra[STEP];                // boundary frame, samples 256..511
    __shared__ float red[256];

    int wg = blockIdx.x;
    int r  = wg >> 3;
    int ft = wg & 7;
    int t0 = ft * 16;
    int tid = threadIdx.x;

    // ---- phase 1: spec rows for frames t0-1 .. t0+15 (t = frame+1 = t0 .. t0+16) ----
    const float* crow = C1 + r * N_COEFFS_PAD;
    for (int k = tid; k < CHUNK; k += 256) {
        float cm = crow[k];
        float cp = crow[CHUNK + k];
        float c0 = crow[2 * CHUNK + k];
        float mag = 0.5f + (1.f / (1.f + __expf(-cm))) * (0.9999f * 0.5f);
        float phi = tanhf(cp) * PI_F;
        float st  = 1.f / (1.f + __expf(-c0));
        float lm  = __logf(mag + 1e-12f);
        bool edge = (k == 0) || (k == 256);
        #pragma unroll 1
        for (int i = 0; i < 17; ++i) {
            int t = t0 + i;                       // frame index = t-1
            float Rv = 0.f, Iv = 0.f;
            if (t > 0) {                          // t==0 => frame -1 => zeros (shift pad)
                float dec = st * __expf(lm * (float)t);
                float th = phi * (float)t;
                th -= 6.28318530718f * rintf(th * 0.15915494309f);  // range-reduce
                float sn, cn;
                __sincosf(th, &sn, &cn);
                Rv = dec * cn;
                Iv = edge ? 0.f : dec * sn;
            }
            S[i][2 * k]     = (f16)Rv;
            S[i][2 * k + 1] = (f16)Iv;
        }
    }
    if (tid < (K2PAD - K2V)) {                    // zero K padding columns
        #pragma unroll 1
        for (int i = 0; i < 17; ++i) S[i][K2V + tid] = (f16)0.f;
    }
    __syncthreads();

    // ---- phase 2: WMMA frame tile (rows 1..16 of S = frames t0..t0+15) ----
    // Wave w owns N-tiles {w, w+8, w+16, w+24}; A fragment shared across all 4.
    int lane = tid & 31, wave = tid >> 5;
    int m = lane & 15, h = lane >> 4;
    {
        const f16* Bbase = Bt + (wave * 16 + m) * K2PAD;   // N-tile q adds 128*K2PAD
        v8f acc0 = {}, acc1 = {}, acc2 = {}, acc3 = {};
        #pragma unroll 1
        for (int kt = 0; kt < 17; ++kt) {
            int kb = kt * 32;
            v16h a;
            a = *(const v16h*)&S[m + 1][kb + 16 * h];
            v16h b0 = *(const v16h*)(Bbase + 0 * 128 * K2PAD + kb + 16 * h);
            v16h b1 = *(const v16h*)(Bbase + 1 * 128 * K2PAD + kb + 16 * h);
            v16h b2 = *(const v16h*)(Bbase + 2 * 128 * K2PAD + kb + 16 * h);
            v16h b3 = *(const v16h*)(Bbase + 3 * 128 * K2PAD + kb + 16 * h);
            acc0 = __builtin_amdgcn_wmma_f32_16x16x32_f16(false, a, false, b0, (short)0, acc0, false, false);
            acc1 = __builtin_amdgcn_wmma_f32_16x16x32_f16(false, a, false, b1, (short)0, acc1, false, false);
            acc2 = __builtin_amdgcn_wmma_f32_16x16x32_f16(false, a, false, b2, (short)0, acc2, false, false);
            acc3 = __builtin_amdgcn_wmma_f32_16x16x32_f16(false, a, false, b3, (short)0, acc3, false, false);
        }
        #pragma unroll
        for (int rr = 0; rr < 8; ++rr) {
            F[rr + 8 * h][(wave +  0) * 16 + m] = acc0[rr];
            F[rr + 8 * h][(wave +  8) * 16 + m] = acc1[rr];
            F[rr + 8 * h][(wave + 16) * 16 + m] = acc2[rr];
            F[rr + 8 * h][(wave + 24) * 16 + m] = acc3[rr];
        }
    }

    // ---- phase 2b: boundary frame (t0-1), second half only, VALU dot on S row 0 ----
    {
        const f16* Brow = Bt + (STEP + tid) * K2PAD;
        float s = 0.f;
        #pragma unroll 8
        for (int k2 = 0; k2 < K2PAD; ++k2) s += (float)S[0][k2] * (float)Brow[k2];
        extra[tid] = s;
    }
    __syncthreads();

    // ---- phase 3: overlap-add + unnormalized store + sumsq partial ----
    float* orow = out + r * OUT_PER_ROW + t0 * STEP;
    float acc2 = 0.f;
    #pragma unroll 1
    for (int f = 0; f < 16; ++f) {
        float v = F[f][tid] + (f == 0 ? extra[tid] : F[f - 1][STEP + tid]);
        orow[f * STEP + tid] = v;
        acc2 += v * v;
    }
    red[tid] = acc2;
    __syncthreads();
    for (int s = 128; s > 0; s >>= 1) {
        if (tid < s) red[tid] += red[tid + s];
        __syncthreads();
    }
    if (tid == 0) partial[wg] = red[0];
}

// ======================= normalization =======================

__global__ void k_norm(const float* __restrict__ partial, float* __restrict__ norms) {
    int r = blockIdx.x * 256 + threadIdx.x;
    if (r < N_ROWS) {
        float s = 0.f;
        #pragma unroll
        for (int i = 0; i < 8; ++i) s += partial[r * 8 + i];   // fixed order: deterministic
        norms[r] = sqrtf(s);
    }
}

__global__ void k_scale(float* __restrict__ out, const float* __restrict__ norms) {
    int i = blockIdx.x * 256 + threadIdx.x;
    out[i] = out[i] / (norms[i >> 15] + 1e-8f);                // 32768 = 2^15 samples/row
}

// ======================= launcher =======================

extern "C" void kernel_launch(void* const* d_in, const int* in_sizes, int n_in,
                              void* d_out, int out_size, void* d_ws, size_t ws_size,
                              hipStream_t stream) {
    const float* sel   = (const float*)d_in[0];   // (8,16,4,512) f32
    const float* items = (const float*)d_in[1];   // (512,771)   f32
    float* out = (float*)d_out;                   // (8,16,4,32768) f32

    char* ws = (char*)d_ws;                       // needs ~3.4 MB
    f16*   A       = (f16*)  (ws + OFF_AF16);
    f16*   It      = (f16*)  (ws + OFF_IT);
    float* C1      = (float*)(ws + OFF_C1);
    f16*   Bt      = (f16*)  (ws + OFF_BT);
    float* partial = (float*)(ws + OFF_PART);
    float* norms   = (float*)(ws + OFF_NORM);

    k_prep_a    <<<1024, 256, 0, stream>>>(sel, A);
    k_prep_items<<<784,  512, 0, stream>>>(items, It);
    k_prep_basis<<<512,  544, 0, stream>>>(Bt);
    k_gemm1     <<<1568, 32,  0, stream>>>(A, It, C1);
    k_synth     <<<4096, 256, 0, stream>>>(C1, Bt, out, partial);
    k_norm      <<<2,    256, 0, stream>>>(partial, norms);
    k_scale     <<<65536,256, 0, stream>>>(out, norms);
}
